// MultiHeadAttention_69947837383448
// MI455X (gfx1250) — matrix operand order repaired and checked by reading, not compiled, not run on hardware
//
#include <hip/hip_runtime.h>
#include <hip/hip_bf16.h>
#include <math.h>

typedef __bf16 bf16;
typedef __attribute__((ext_vector_type(16))) __bf16 v16bf;
typedef __attribute__((ext_vector_type(8)))  __bf16 v8bf;
typedef __attribute__((ext_vector_type(4)))  __bf16 v4bf;
typedef __attribute__((ext_vector_type(8)))  float  v8f;
typedef __attribute__((ext_vector_type(4)))  float  v4f;

#define DIMM 768
#define QKV_OUT 2304
#define NSEQ 2048
#define NHEAD 16
#define DK 48
#define BATCH 4

union V16 { v16bf v; v8bf h[2]; };

#define WMMA_BF16(A, B, C) \
  __builtin_amdgcn_wmma_f32_16x16x32_bf16(false, (A), false, (B), (short)0, (C), false, false)

__device__ inline v8bf zero8() {
  v8bf z;
#pragma unroll
  for (int i = 0; i < 8; ++i) z[i] = (bf16)0.0f;
  return z;
}
__device__ inline v16bf zero16() {
  v16bf z;
#pragma unroll
  for (int i = 0; i < 16; ++i) z[i] = (bf16)0.0f;
  return z;
}

// ---------------------------------------------------------------- f32 -> bf16
__global__ void cvt_f32_bf16(const float* __restrict__ in, bf16* __restrict__ out, int n) {
  int i = (blockIdx.x * blockDim.x + threadIdx.x) * 4;
  if (i + 3 < n) {
    v4f f = *(const v4f*)(in + i);
    v4bf o;
    o[0] = (bf16)f[0]; o[1] = (bf16)f[1]; o[2] = (bf16)f[2]; o[3] = (bf16)f[3];
    *(v4bf*)(out + i) = o;
  }
}

// ------------------------------------------------------------------ QKV GEMM
// C[8192 x 2304] = Xbf[8192x768] @ Wbf^T (+bias). Wave tile: 32 rows x 64 cols.
// Scatters into:
//   qbuf[bh][n][48]  (pre-scaled by 1/sqrt(48))
//   kbuf[bh][n][48]
//   vT  [bh][48][2048]
__global__ __launch_bounds__(128) void qkv_gemm(
    const bf16* __restrict__ X, const bf16* __restrict__ W, const float* __restrict__ bias,
    bf16* __restrict__ qbuf, bf16* __restrict__ kbuf, bf16* __restrict__ vT) {
  const int lane = threadIdx.x & 31;
  const int wave = threadIdx.x >> 5;
  const int widx = blockIdx.x * 4 + wave;
  const int NT = QKV_OUT / 64;      // 36 column tiles of 64
  const int mTile = widx / NT;      // 0..255 (32-row tiles)
  const int nTile = widx % NT;
  const int m0 = mTile * 32;
  const int n0 = nTile * 64;
  const int c  = lane & 15;
  const int hi = lane >> 4;

  v8f acc[2][4] = {};
  const bf16* ar0 = X + (m0 + c) * DIMM;
  const bf16* ar1 = X + (m0 + 16 + c) * DIMM;
  const bf16* wr0 = W + (n0 +  0 + c) * DIMM + 16 * hi;
  const bf16* wr1 = W + (n0 + 16 + c) * DIMM + 16 * hi;
  const bf16* wr2 = W + (n0 + 32 + c) * DIMM + 16 * hi;
  const bf16* wr3 = W + (n0 + 48 + c) * DIMM + 16 * hi;

  for (int k = 0; k < DIMM; k += 32) {
    // warm next chunk into WGP$/L2 (global_prefetch_b8)
    if (k + 32 < DIMM) {
      __builtin_prefetch(ar0 + k + 32, 0, 3);
      __builtin_prefetch(ar1 + k + 32, 0, 3);
      __builtin_prefetch(wr0 + k + 32, 0, 3);
      __builtin_prefetch(wr2 + k + 32, 0, 3);
    }
    V16 a0, a1;
    a0.h[0] = *(const v8bf*)(ar0 + k + 16 * hi);
    a0.h[1] = *(const v8bf*)(ar0 + k + 16 * hi + 8);
    a1.h[0] = *(const v8bf*)(ar1 + k + 16 * hi);
    a1.h[1] = *(const v8bf*)(ar1 + k + 16 * hi + 8);
    v16bf b0 = *(const v16bf*)(wr0 + k);
    v16bf b1 = *(const v16bf*)(wr1 + k);
    v16bf b2 = *(const v16bf*)(wr2 + k);
    v16bf b3 = *(const v16bf*)(wr3 + k);
    acc[0][0] = WMMA_BF16(a0.v, b0, acc[0][0]);
    acc[0][1] = WMMA_BF16(a0.v, b1, acc[0][1]);
    acc[0][2] = WMMA_BF16(a0.v, b2, acc[0][2]);
    acc[0][3] = WMMA_BF16(a0.v, b3, acc[0][3]);
    acc[1][0] = WMMA_BF16(a1.v, b0, acc[1][0]);
    acc[1][1] = WMMA_BF16(a1.v, b1, acc[1][1]);
    acc[1][2] = WMMA_BF16(a1.v, b2, acc[1][2]);
    acc[1][3] = WMMA_BF16(a1.v, b3, acc[1][3]);
  }

  const int bIdx = m0 >> 11;          // batch index (32-row tile never crosses batch)
  const int nRow = m0 & (NSEQ - 1);   // sequence pos of row 0
#pragma unroll
  for (int s = 0; s < 4; ++s) {
    const int e     = n0 + s * 16 + c;
    const int which = e / DIMM;              // 0=q 1=k 2=v
    const int rem   = e - which * DIMM;
    const int head  = rem / DK;
    const int d     = rem - head * DK;
    const int bh    = bIdx * NHEAD + head;
    const float bv  = bias[e];
#pragma unroll
    for (int r = 0; r < 2; ++r) {
#pragma unroll
      for (int j = 0; j < 8; ++j) {
        const int n = nRow + 16 * r + j + 8 * hi;
        const float v = acc[r][s][j] + bv;
        if (which == 0)      qbuf[(bh * NSEQ + n) * DK + d] = (bf16)(v * 0.14433756729740643f);
        else if (which == 1) kbuf[(bh * NSEQ + n) * DK + d] = (bf16)v;
        else                 vT[(bh * DK + d) * NSEQ + n]   = (bf16)v;
      }
    }
  }
}

// ------------------------------------------------------------- Flash attention
// One wave per (bh, 16-query tile). d_k padded 48 -> 64 for the score WMMAs.
__global__ __launch_bounds__(32) void flash_attn(
    const bf16* __restrict__ qbuf, const bf16* __restrict__ kbuf,
    const bf16* __restrict__ vT, bf16* __restrict__ attn) {
  __shared__ __attribute__((aligned(32))) bf16 pTile[16 * 40];  // P bounce, stride 40

  const int lane = threadIdx.x;
  const int c  = lane & 15;
  const int hi = lane >> 4;
  const int bh    = blockIdx.x >> 7;     // /128
  const int qTile = blockIdx.x & 127;
  const int q0 = qTile * 16;

  // Q as A-operand: rows = lane&15, two K-chunks (0..31, 32..63 zero-padded)
  const bf16* qr = qbuf + (bh * NSEQ + q0 + c) * DK;
  V16 qa0, qa1;
  qa0.h[0] = *(const v8bf*)(qr + 16 * hi);
  qa0.h[1] = *(const v8bf*)(qr + 16 * hi + 8);
  qa1.h[0] = hi == 0 ? *(const v8bf*)(qr + 32) : zero8();   // hi0: d = 32..39; hi1: d = 48..55 pad
  qa1.h[1] = hi == 0 ? *(const v8bf*)(qr + 40) : zero8();   // hi0: d = 40..47; hi1: d = 56..63 pad

  float mprev[8], lsum[8];
#pragma unroll
  for (int j = 0; j < 8; ++j) { mprev[j] = -__builtin_huge_valf(); lsum[j] = 0.0f; }
  v8f o0 = {}, o1 = {}, o2 = {};

  for (int j0 = 0; j0 < NSEQ; j0 += 32) {
    // ---- scores: S[16 x 32] = Q @ K^T (two 16-col sub-tiles)
    v8f s0 = {}, s1 = {};
#pragma unroll
    for (int t = 0; t < 2; ++t) {
      const bf16* kr = kbuf + (bh * NSEQ + j0 + t * 16 + c) * DK;
      v16bf kb0 = *(const v16bf*)(kr + 16 * hi);       // d = 16hi .. 16hi+15
      v16bf kb1;
      if (hi == 0) kb1 = *(const v16bf*)(kr + 32);     // d = 32..47
      else         kb1 = zero16();                     // d = 48..63 pad
      v8f sv = {};
      sv = WMMA_BF16(qa0.v, kb0, sv);
      sv = WMMA_BF16(qa1.v, kb1, sv);
      if (t == 0) s0 = sv; else s1 = sv;
    }

    // ---- online softmax; each 16-lane half owns rows j+8*hi
#pragma unroll
    for (int j = 0; j < 8; ++j) {
      float a = s0[j], b = s1[j];
      float mx = fmaxf(a, b);
      mx = fmaxf(mx, __shfl_xor(mx, 1, 32));
      mx = fmaxf(mx, __shfl_xor(mx, 2, 32));
      mx = fmaxf(mx, __shfl_xor(mx, 4, 32));
      mx = fmaxf(mx, __shfl_xor(mx, 8, 32));
      float mnew = fmaxf(mprev[j], mx);
      float sc = __expf(mprev[j] - mnew);
      float p0 = __expf(a - mnew);
      float p1 = __expf(b - mnew);
      float rs = p0 + p1;
      rs += __shfl_xor(rs, 1, 32);
      rs += __shfl_xor(rs, 2, 32);
      rs += __shfl_xor(rs, 4, 32);
      rs += __shfl_xor(rs, 8, 32);
      lsum[j] = lsum[j] * sc + rs;
      mprev[j] = mnew;
      o0[j] *= sc; o1[j] *= sc; o2[j] *= sc;
      const int row = j + 8 * hi;
      pTile[row * 40 + c]      = (bf16)p0;
      pTile[row * 40 + 16 + c] = (bf16)p1;
    }
    __syncthreads();

    // ---- O += P @ V  (P re-read from LDS in A-operand layout)
    V16 pa;
    pa.h[0] = *(const v8bf*)(&pTile[c * 40 + 16 * hi]);
    pa.h[1] = *(const v8bf*)(&pTile[c * 40 + 16 * hi + 8]);
    const bf16* vr = vT + (bh * DK + c) * NSEQ + j0 + 16 * hi;
    v16bf vb0 = *(const v16bf*)(vr);
    v16bf vb1 = *(const v16bf*)(vr + 16 * NSEQ);
    v16bf vb2 = *(const v16bf*)(vr + 32 * NSEQ);
    o0 = WMMA_BF16(pa.v, vb0, o0);
    o1 = WMMA_BF16(pa.v, vb1, o1);
    o2 = WMMA_BF16(pa.v, vb2, o2);
    __syncthreads();
  }

  // ---- finalize and write attn[b][n][head*48 + d] (bf16, feeds out-proj GEMM)
  const int b = bh >> 4, head = bh & 15;
#pragma unroll
  for (int j = 0; j < 8; ++j) {
    const float inv = 1.0f / lsum[j];
    const int n = q0 + j + 8 * hi;
    bf16* orow = attn + (b * NSEQ + n) * DIMM + head * DK;
    orow[c]      = (bf16)(o0[j] * inv);
    orow[16 + c] = (bf16)(o1[j] * inv);
    orow[32 + c] = (bf16)(o2[j] * inv);
  }
}

// ----------------------------------------------------------- output projection
// Wave tile: 32 rows x 64 cols, fp32 output + bias.
__global__ __launch_bounds__(128) void out_gemm(
    const bf16* __restrict__ A, const bf16* __restrict__ W, const float* __restrict__ bias,
    float* __restrict__ out) {
  const int lane = threadIdx.x & 31;
  const int wave = threadIdx.x >> 5;
  const int widx = blockIdx.x * 4 + wave;
  const int NT = DIMM / 64;     // 12 column tiles
  const int mTile = widx / NT;  // 0..255
  const int nTile = widx % NT;
  const int m0 = mTile * 32;
  const int n0 = nTile * 64;
  const int c  = lane & 15;
  const int hi = lane >> 4;

  v8f acc[2][4] = {};
  const bf16* ar0 = A + (m0 + c) * DIMM;
  const bf16* ar1 = A + (m0 + 16 + c) * DIMM;
  const bf16* wr0 = W + (n0 +  0 + c) * DIMM + 16 * hi;
  const bf16* wr1 = W + (n0 + 16 + c) * DIMM + 16 * hi;
  const bf16* wr2 = W + (n0 + 32 + c) * DIMM + 16 * hi;
  const bf16* wr3 = W + (n0 + 48 + c) * DIMM + 16 * hi;

  for (int k = 0; k < DIMM; k += 32) {
    if (k + 32 < DIMM) {
      __builtin_prefetch(ar0 + k + 32, 0, 3);
      __builtin_prefetch(ar1 + k + 32, 0, 3);
      __builtin_prefetch(wr0 + k + 32, 0, 3);
      __builtin_prefetch(wr2 + k + 32, 0, 3);
    }
    V16 a0, a1;
    a0.h[0] = *(const v8bf*)(ar0 + k + 16 * hi);
    a0.h[1] = *(const v8bf*)(ar0 + k + 16 * hi + 8);
    a1.h[0] = *(const v8bf*)(ar1 + k + 16 * hi);
    a1.h[1] = *(const v8bf*)(ar1 + k + 16 * hi + 8);
    v16bf b0 = *(const v16bf*)(wr0 + k);
    v16bf b1 = *(const v16bf*)(wr1 + k);
    v16bf b2 = *(const v16bf*)(wr2 + k);
    v16bf b3 = *(const v16bf*)(wr3 + k);
    acc[0][0] = WMMA_BF16(a0.v, b0, acc[0][0]);
    acc[0][1] = WMMA_BF16(a0.v, b1, acc[0][1]);
    acc[0][2] = WMMA_BF16(a0.v, b2, acc[0][2]);
    acc[0][3] = WMMA_BF16(a0.v, b3, acc[0][3]);
    acc[1][0] = WMMA_BF16(a1.v, b0, acc[1][0]);
    acc[1][1] = WMMA_BF16(a1.v, b1, acc[1][1]);
    acc[1][2] = WMMA_BF16(a1.v, b2, acc[1][2]);
    acc[1][3] = WMMA_BF16(a1.v, b3, acc[1][3]);
  }
#pragma unroll
  for (int s = 0; s < 4; ++s) {
    const int e = n0 + s * 16 + c;
    const float bv = bias[e];
#pragma unroll
    for (int r = 0; r < 2; ++r) {
#pragma unroll
      for (int j = 0; j < 8; ++j) {
        out[(m0 + 16 * r + j + 8 * hi) * DIMM + e] = acc[r][s][j] + bv;
      }
    }
  }
}

// ----------------------------------------------------------------------- host
extern "C" void kernel_launch(void* const* d_in, const int* in_sizes, int n_in,
                              void* d_out, int out_size, void* d_ws, size_t ws_size,
                              hipStream_t stream) {
  (void)in_sizes; (void)n_in; (void)out_size; (void)ws_size;
  const float* x     = (const float*)d_in[0];
  const float* qkv_w = (const float*)d_in[1];
  const float* qkv_b = (const float*)d_in[2];
  const float* out_w = (const float*)d_in[3];
  const float* out_b = (const float*)d_in[4];
  float* out = (float*)d_out;

  const int M = BATCH * NSEQ;                 // 8192
  char* ws = (char*)d_ws;
  bf16* x_bf  = (bf16*)ws;  ws += (size_t)M * DIMM * 2;          // 12.6 MB
  bf16* w1_bf = (bf16*)ws;  ws += (size_t)QKV_OUT * DIMM * 2;    //  3.5 MB
  bf16* w2_bf = (bf16*)ws;  ws += (size_t)DIMM * DIMM * 2;       //  1.2 MB
  bf16* qbuf  = (bf16*)ws;  ws += (size_t)M * DIMM * 2;          // 12.6 MB
  bf16* kbuf  = (bf16*)ws;  ws += (size_t)M * DIMM * 2;          // 12.6 MB
  bf16* vT    = (bf16*)ws;  ws += (size_t)M * DIMM * 2;          // 12.6 MB
  bf16* attn  = (bf16*)ws;  ws += (size_t)M * DIMM * 2;          // 12.6 MB

  const int nx = M * DIMM, nw1 = QKV_OUT * DIMM, nw2 = DIMM * DIMM;
  cvt_f32_bf16<<<(nx  / 4 + 255) / 256, 256, 0, stream>>>(x, x_bf, nx);
  cvt_f32_bf16<<<(nw1 / 4 + 255) / 256, 256, 0, stream>>>(qkv_w, w1_bf, nw1);
  cvt_f32_bf16<<<(nw2 / 4 + 255) / 256, 256, 0, stream>>>(out_w, w2_bf, nw2);

  // 256 row-tiles (32 rows) x 36 col-tiles, 4 waves/block
  qkv_gemm<<<(M / 32) * (QKV_OUT / 64) / 4, 128, 0, stream>>>(x_bf, w1_bf, qkv_b,
                                                              qbuf, kbuf, vT);
  // one wave per (b,h, 16-query tile): 64 * 128 blocks
  flash_attn<<<BATCH * NHEAD * (NSEQ / 16), 32, 0, stream>>>(qbuf, kbuf, vT, attn);
  // 256 x 12 tiles, 4 waves/block
  out_gemm<<<(M / 32) * (DIMM / 64) / 4, 128, 0, stream>>>(attn, w2_bf, out_b, out);
}
